// SimpleRNN_55662776156330
// MI455X (gfx1250) — matrix operand order repaired and checked by reading, not compiled, not run on hardware
//
#include <hip/hip_runtime.h>
#include <hip/hip_bf16.h>

// ---------------------------------------------------------------------------
// SimpleRNN on MI455X (gfx1250): persistent per-batch-tile recurrence with
// bf16 WMMA (v_wmma_f32_16x16x32_bf16), f32 accumulation, tanh via v_tanh_f32.
// Round 2: double-buffered LDS hidden state (1 barrier/step) + LDS-staged x.
// ---------------------------------------------------------------------------

constexpr int kB   = 512;   // batch
constexpr int kS   = 512;   // timesteps
constexpr int kH   = 1024;  // hidden
constexpr int kOUT = 256;   // output
constexpr int kTileM = 16;            // batch rows per workgroup
constexpr int kWaves = 16;            // waves per workgroup (512 threads)
constexpr int kLdsH  = kH + 8;        // padded row stride (bf16 elems) -> bank spread
constexpr int kHBuf  = kTileM * kLdsH; // one h buffer, in bf16 elements

typedef __attribute__((ext_vector_type(16))) __bf16 v16bf;
typedef __attribute__((ext_vector_type(8)))  float  v8f;

struct B32x8 { uint4 lo, hi; };   // 32 bytes == one v16bf fragment

#if __has_builtin(__builtin_amdgcn_tanhf)
__device__ __forceinline__ float fast_tanh(float x) { return __builtin_amdgcn_tanhf(x); }
#else
__device__ __forceinline__ float fast_tanh(float x) { return tanhf(x); }
#endif

// A-fragment (16x32 bf16, M x K) from LDS.
// ISA layout: lanes 0-15 -> row M=lane, K = kb+{0..7} and kb+{16..23}
//             lanes16-31 -> row M=lane-16, K = kb+{8..15} and kb+{24..31}
__device__ __forceinline__ v16bf load_a_frag(const __bf16* hbuf, int lane, int kb) {
    const int row  = lane & 15;
    const int half = lane >> 4;
    const __bf16* p = hbuf + row * kLdsH + kb + half * 8;
    B32x8 u;
    u.lo = *(const uint4*)(p);        // 8 bf16  (K base)
    u.hi = *(const uint4*)(p + 16);   // 8 bf16  (K base + 16)
    return __builtin_bit_cast(v16bf, u);
}

// B-fragment (32x16 bf16, K x N) from row-major weight W[N][kH] (B = W^T slice).
// ISA layout: lanes 0-15 -> col N=lane, K = kb+{0..7} and kb+{16..23} ; lanes16-31 -> col N=lane-16, K = kb+{8..15} and kb+{24..31}
__device__ __forceinline__ v16bf load_b_frag(const __bf16* __restrict__ W, int n0, int kb, int lane) {
    const int col  = n0 + (lane & 15);
    const int half = lane >> 4;
    const __bf16* p = W + (size_t)col * kH + kb + half * 8;
    B32x8 u;
    u.lo = *(const uint4*)(p);        // 8 bf16  (K base)
    u.hi = *(const uint4*)(p + 16);   // 8 bf16  (K base + 16)
    return __builtin_bit_cast(v16bf, u);
}

// ---------------------------------------------------------------------------
// Prologue: f32 -> bf16 weight conversion + fused bias.
// ---------------------------------------------------------------------------
__global__ void rnn_cvt_weights(const float* __restrict__ Wh, const float* __restrict__ Wo,
                                const float* __restrict__ bi, const float* __restrict__ bh,
                                __bf16* __restrict__ Wh_bf, __bf16* __restrict__ Wo_bf,
                                float* __restrict__ biasSum) {
    const int tid    = blockIdx.x * blockDim.x + threadIdx.x;
    const int stride = gridDim.x * blockDim.x;
    for (int i = tid; i < kH * kH; i += stride)   Wh_bf[i] = (__bf16)Wh[i];
    for (int i = tid; i < kOUT * kH; i += stride) Wo_bf[i] = (__bf16)Wo[i];
    for (int i = tid; i < kH; i += stride)        biasSum[i] = bi[i] + bh[i];
}

// ---------------------------------------------------------------------------
// Persistent RNN kernel: one workgroup per 16 batch rows, full 512-step scan.
// ---------------------------------------------------------------------------
__global__ __launch_bounds__(kWaves * 32, 1)
void rnn_persistent(const float* __restrict__ x,        // [B, S] f32
                    const float* __restrict__ Wi,       // [H]    f32 (IN == 1)
                    const __bf16* __restrict__ Wh_bf,   // [H, H] bf16
                    const __bf16* __restrict__ Wo_bf,   // [OUT,H] bf16
                    const float* __restrict__ biasSum,  // [H]    f32 (bi+bh)
                    const float* __restrict__ bo,       // [OUT]  f32
                    float* __restrict__ out)            // [B, OUT] f32
{
    __shared__ __bf16 h_lds[2 * kHBuf];        // double-buffered hidden state tile
    __shared__ float  x_lds[kS * kTileM];      // x transposed: x_lds[t*16 + row]
    __shared__ float  wi_s[kH];
    __shared__ float  bs_s[kH];

    const int tid  = threadIdx.x;
    const int wave = tid >> 5;
    const int lane = tid & 31;
    const int wg   = blockIdx.x;          // batch rows [wg*16, wg*16+16)

    // h0 = 0 (both buffers); stage per-column constants + transposed x in LDS
    for (int i = tid; i < 2 * kHBuf; i += kWaves * 32) h_lds[i] = (__bf16)0.0f;
    for (int i = tid; i < kH; i += kWaves * 32) { wi_s[i] = Wi[i]; bs_s[i] = biasSum[i]; }
    for (int i = tid; i < kS * kTileM; i += kWaves * 32) {
        const int r = i & (kTileM - 1);
        const int t = i >> 4;
        x_lds[i] = x[(size_t)(wg * kTileM + r) * kS + t];
    }
    __syncthreads();

    const int nl   = lane & 15;
    const int half = lane >> 4;
    const int row0 = half * 8;            // C/D layout: VGPR r holds M = r + 8*half
    const int nb   = wave * 64;           // this wave's 64-column slice of H

    int rd = 0, wr = kHBuf;               // double-buffer offsets

    for (int t = 0; t < kS; ++t) {
        v8f acc[4] = {v8f{}, v8f{}, v8f{}, v8f{}};

        // GEMM: h_new_tile[16, 64] += h_tile[16,1024] @ Wh^T[1024, 64]
        for (int kb = 0; kb < kH; kb += 32) {
            const v16bf a = load_a_frag(h_lds + rd, lane, kb);
#pragma unroll
            for (int nt = 0; nt < 4; ++nt) {
                const v16bf b = load_b_frag(Wh_bf, nb + nt * 16, kb, lane);
                acc[nt] = __builtin_amdgcn_wmma_f32_16x16x32_bf16(
                    false, a, false, b, (short)0, acc[nt], false, false);
            }
        }

        // epilogue: + x_t*Wi + (bi+bh), tanh, write bf16 h_new into the OTHER buffer
        const float4 xa = *(const float4*)&x_lds[t * kTileM + row0];
        const float4 xb = *(const float4*)&x_lds[t * kTileM + row0 + 4];
        const float xr[8] = {xa.x, xa.y, xa.z, xa.w, xb.x, xb.y, xb.z, xb.w};

#pragma unroll
        for (int nt = 0; nt < 4; ++nt) {
            const int col  = nb + nt * 16 + nl;
            const float wic = wi_s[col];
            const float bsc = bs_s[col];
#pragma unroll
            for (int r = 0; r < 8; ++r) {
                const float v = fast_tanh(acc[nt][r] + xr[r] * wic + bsc);
                h_lds[wr + (row0 + r) * kLdsH + col] = (__bf16)v;
            }
        }
        __syncthreads();   // writes of step t done before reads of step t+1
        const int tmp = rd; rd = wr; wr = tmp;
    }

    // Output GEMM: out[16, 256] = h_T[16,1024] @ Wo^T + bo   (1 tile per wave)
    {
        v8f acc = v8f{};
        const int n0 = wave * 16;         // 16 waves * 16 = 256 = OUT
        for (int kb = 0; kb < kH; kb += 32) {
            const v16bf a = load_a_frag(h_lds + rd, lane, kb);
            const v16bf b = load_b_frag(Wo_bf, n0, kb, lane);
            acc = __builtin_amdgcn_wmma_f32_16x16x32_bf16(
                false, a, false, b, (short)0, acc, false, false);
        }
        const int col  = n0 + nl;
        const float bc = bo[col];
#pragma unroll
        for (int r = 0; r < 8; ++r)
            out[(size_t)(wg * kTileM + row0 + r) * kOUT + col] = acc[r] + bc;
    }
}

// ---------------------------------------------------------------------------
extern "C" void kernel_launch(void* const* d_in, const int* in_sizes, int n_in,
                              void* d_out, int out_size, void* d_ws, size_t ws_size,
                              hipStream_t stream) {
    const float* x  = (const float*)d_in[0];   // [B, S]
    const float* Wi = (const float*)d_in[1];   // [H, 1]
    const float* bi = (const float*)d_in[2];   // [H]
    const float* Wh = (const float*)d_in[3];   // [H, H]
    const float* bh = (const float*)d_in[4];   // [H]
    const float* Wo = (const float*)d_in[5];   // [OUT, H]
    const float* bo = (const float*)d_in[6];   // [OUT]
    float* out = (float*)d_out;

    char* ws = (char*)d_ws;
    __bf16* Wh_bf   = (__bf16*)ws;                                   // 2 MB
    __bf16* Wo_bf   = (__bf16*)(ws + (size_t)kH * kH * 2);           // 512 KB
    float*  biasSum = (float*)(ws + (size_t)kH * kH * 2 + (size_t)kOUT * kH * 2); // 4 KB

    rnn_cvt_weights<<<512, 256, 0, stream>>>(Wh, Wo, bi, bh, Wh_bf, Wo_bf, biasSum);
    rnn_persistent<<<kB / kTileM, kWaves * 32, 0, stream>>>(
        x, Wi, Wh_bf, Wo_bf, biasSum, bo, out);
}
